// RGNN_Classifier_21766894256131
// MI455X (gfx1250) — matrix operand order repaired and checked by reading, not compiled, not run on hardware
//
#include <hip/hip_runtime.h>
#include <hip/hip_bf16.h>

typedef __attribute__((ext_vector_type(16))) __bf16 v16bf;
typedef __attribute__((ext_vector_type(8)))  float  v8f;

#define NN     50000
#define EE     800000
#define HID    256
#define NGRAPH 64
#define NREL   3

__device__ __forceinline__ float4 ldg4(const float* p){ return *(const float4*)p; }

// ---------------- weight transpose + bf16 convert ----------------
// W: [K][256] row-major (f32)  ->  Wt[n*ldt + koff + k] = bf16(W[k][n])
__global__ void k_transpose_bf16(const float* __restrict__ W, __bf16* __restrict__ Wt,
                                 int K, int ldt, int koff){
  int i = blockIdx.x * blockDim.x + threadIdx.x;
  if (i >= K * HID) return;
  int k = i >> 8, n = i & 255;
  Wt[(size_t)n * ldt + koff + k] = (__bf16)W[(size_t)k * HID + n];
}

// ---------------- WMMA GEMM: C[M,256] (+)= A[M,K] @ W + bias ----------------
// One wave computes a 16x64 strip: 4 accumulators share one A fragment per
// k-step (A loaded + converted to bf16 once, reused by 4 v_wmma ops).
__global__ void k_wmma_gemm(const float* __restrict__ A, int lda, int K,
                            const __bf16* __restrict__ Wt,
                            const float* __restrict__ bias,
                            float* __restrict__ C, int M, int accum, int relu){
  const int wave = threadIdx.x >> 5;
  const int lane = threadIdx.x & 31;
  const int wid  = blockIdx.x * 8 + wave;
  const int mt = wid >> 2, ng = wid & 3;       // 4 N-groups of 64 (N == 256)
  const int m0 = mt * 16, n0 = ng * 64;
  if (m0 >= M) return;
  const int half = lane >> 4, l15 = lane & 15;
  const float*  arow = A + (size_t)(m0 + l15) * lda;
  const __bf16* b0 = Wt + (size_t)(n0 +  0 + l15) * K;
  const __bf16* b1 = Wt + (size_t)(n0 + 16 + l15) * K;
  const __bf16* b2 = Wt + (size_t)(n0 + 32 + l15) * K;
  const __bf16* b3 = Wt + (size_t)(n0 + 48 + l15) * K;
  v8f c0 = {}, c1 = {}, c2 = {}, c3 = {};
  for (int k = 0; k < K; k += 32){
    const int ka = k + half * 16;              // A frag: K {ka..ka+15}
    float4 f0 = ldg4(arow + ka);
    float4 f1 = ldg4(arow + ka + 4);
    float4 f2 = ldg4(arow + ka + 8);
    float4 f3 = ldg4(arow + ka + 12);
    v16bf a;
    a[0]=(__bf16)f0.x;  a[1]=(__bf16)f0.y;  a[2]=(__bf16)f0.z;  a[3]=(__bf16)f0.w;
    a[4]=(__bf16)f1.x;  a[5]=(__bf16)f1.y;  a[6]=(__bf16)f1.z;  a[7]=(__bf16)f1.w;
    a[8]=(__bf16)f2.x;  a[9]=(__bf16)f2.y;  a[10]=(__bf16)f2.z; a[11]=(__bf16)f2.w;
    a[12]=(__bf16)f3.x; a[13]=(__bf16)f3.y; a[14]=(__bf16)f3.z; a[15]=(__bf16)f3.w;
    const int kb = k + half * 16;              // B frag: contiguous 16 bf16
    v16bf w0 = *(const v16bf*)(b0 + kb);
    v16bf w1 = *(const v16bf*)(b1 + kb);
    v16bf w2 = *(const v16bf*)(b2 + kb);
    v16bf w3 = *(const v16bf*)(b3 + kb);
    c0 = __builtin_amdgcn_wmma_f32_16x16x32_bf16(false, a, false, w0, (short)0, c0, false, false);
    c1 = __builtin_amdgcn_wmma_f32_16x16x32_bf16(false, a, false, w1, (short)0, c1, false, false);
    c2 = __builtin_amdgcn_wmma_f32_16x16x32_bf16(false, a, false, w2, (short)0, c2, false, false);
    c3 = __builtin_amdgcn_wmma_f32_16x16x32_bf16(false, a, false, w3, (short)0, c3, false, false);
  }
  #pragma unroll
  for (int j = 0; j < 4; ++j){
    v8f cj = (j == 0) ? c0 : (j == 1) ? c1 : (j == 2) ? c2 : c3;
    const int nj = n0 + j * 16 + l15;
    const float bs = bias ? bias[nj] : 0.f;
    float* crow = C + (size_t)(m0 + half * 8) * HID + nj;
    #pragma unroll
    for (int r = 0; r < 8; ++r){
      float v = cj[r] + bs;
      if (relu)  v = fmaxf(v, 0.f);
      if (accum) crow[(size_t)r * HID] += v;
      else       crow[(size_t)r * HID]  = v;
    }
  }
}

// ---------------- edge scatter: acc[dst][rel*256+c] += h[src][c]; cnt[rel][dst]++ ----------------
__global__ void k_scatter(const float* __restrict__ h, const int* __restrict__ src,
                          const int* __restrict__ dst, const int* __restrict__ et,
                          float* __restrict__ acc, float* __restrict__ cnt){
  const int wave = threadIdx.x >> 5, lane = threadIdx.x & 31;
  const int e = blockIdx.x * 8 + wave;
  if (e >= EE) return;
  const int s = src[e], d = dst[e], r = et[e];
  const float* hp = h + (size_t)s * HID + lane * 8;
  float4 a = ldg4(hp), b = ldg4(hp + 4);
  float* ap = acc + (size_t)d * (NREL * HID) + r * HID + lane * 8;
  atomicAdd(ap + 0, a.x); atomicAdd(ap + 1, a.y); atomicAdd(ap + 2, a.z); atomicAdd(ap + 3, a.w);
  atomicAdd(ap + 4, b.x); atomicAdd(ap + 5, b.y); atomicAdd(ap + 6, b.z); atomicAdd(ap + 7, b.w);
  if (lane == 0) atomicAdd(cnt + (size_t)r * NN + d, 1.0f);
}

// ---------------- mean: acc /= max(cnt,1) ----------------
__global__ void k_mean(float* __restrict__ acc, const float* __restrict__ cnt){
  int i = blockIdx.x * blockDim.x + threadIdx.x;       // float4 index
  if (i >= NN * NREL * HID / 4) return;
  int e0 = i * 4;
  int n  = e0 / (NREL * HID);
  int w  = e0 - n * (NREL * HID);
  int r  = w >> 8;
  float inv = 1.0f / fmaxf(cnt[(size_t)r * NN + n], 1.0f);
  float4* p = (float4*)acc + i;
  float4 v = *p;
  v.x *= inv; v.y *= inv; v.z *= inv; v.w *= inv;
  *p = v;
}

// ---------------- fused residual + ReLU + LayerNorm (one wave per row) ----------------
__global__ void k_epilogue(const float* __restrict__ gout, float* __restrict__ h,
                           const float* __restrict__ lnw, const float* __restrict__ lnb){
  const int wave = threadIdx.x >> 5, lane = threadIdx.x & 31;
  const int row = blockIdx.x * 8 + wave;
  if (row >= NN) return;
  const float* gp = gout + (size_t)row * HID + lane * 8;
  float*       hp = h    + (size_t)row * HID + lane * 8;
  float4 g0 = ldg4(gp), g1 = ldg4(gp + 4);
  float4 h0 = ldg4(hp), h1 = ldg4(hp + 4);
  float v[8] = { fmaxf(g0.x + h0.x, 0.f), fmaxf(g0.y + h0.y, 0.f),
                 fmaxf(g0.z + h0.z, 0.f), fmaxf(g0.w + h0.w, 0.f),
                 fmaxf(g1.x + h1.x, 0.f), fmaxf(g1.y + h1.y, 0.f),
                 fmaxf(g1.z + h1.z, 0.f), fmaxf(g1.w + h1.w, 0.f) };
  float s = 0.f;
  #pragma unroll
  for (int i = 0; i < 8; ++i) s += v[i];
  #pragma unroll
  for (int off = 16; off > 0; off >>= 1) s += __shfl_xor(s, off, 32);
  const float mu = s * (1.0f / HID);
  float var = 0.f;
  #pragma unroll
  for (int i = 0; i < 8; ++i){ float d = v[i] - mu; var += d * d; }
  #pragma unroll
  for (int off = 16; off > 0; off >>= 1) var += __shfl_xor(var, off, 32);
  const float inv = rsqrtf(var * (1.0f / HID) + 1e-5f);
  const int c0 = lane * 8;
  float o[8];
  #pragma unroll
  for (int i = 0; i < 8; ++i) o[i] = (v[i] - mu) * inv * lnw[c0 + i] + lnb[c0 + i];
  *(float4*)hp       = make_float4(o[0], o[1], o[2], o[3]);
  *(float4*)(hp + 4) = make_float4(o[4], o[5], o[6], o[7]);
}

// ---------------- global max pool ----------------
__global__ void k_pool_init(float* pooled){
  int i = blockIdx.x * blockDim.x + threadIdx.x;
  if (i < NGRAPH * HID) pooled[i] = -__builtin_inff();
}
__device__ __forceinline__ void atomicMaxF(float* a, float v){
  if (v >= 0.f) atomicMax((int*)a, __float_as_int(v));
  else          atomicMin((unsigned int*)a, __float_as_uint(v));
}
__global__ void k_pool(const float* __restrict__ h, const int* __restrict__ batch,
                       float* __restrict__ pooled){
  const int wave = threadIdx.x >> 5, lane = threadIdx.x & 31;
  const int n = blockIdx.x * 8 + wave;
  if (n >= NN) return;
  const int g = batch[n];
  const float* hp = h + (size_t)n * HID + lane * 8;
  float4 a = ldg4(hp), b = ldg4(hp + 4);
  float* pp = pooled + (size_t)g * HID + lane * 8;
  atomicMaxF(pp + 0, a.x); atomicMaxF(pp + 1, a.y); atomicMaxF(pp + 2, a.z); atomicMaxF(pp + 3, a.w);
  atomicMaxF(pp + 4, b.x); atomicMaxF(pp + 5, b.y); atomicMaxF(pp + 6, b.z); atomicMaxF(pp + 7, b.w);
}

// ---------------- final N=4 projection (scalar) ----------------
__global__ void k_cls2(const float* __restrict__ z, const float* __restrict__ w2,
                       const float* __restrict__ b2, float* __restrict__ out){
  int i = blockIdx.x * blockDim.x + threadIdx.x;
  if (i >= NGRAPH * 4) return;
  int gg = i >> 2, o = i & 3;
  const float* zr = z + (size_t)gg * HID;
  float s = b2[o];
  for (int k = 0; k < HID; ++k) s = fmaf(zr[k], w2[(size_t)k * 4 + o], s);
  out[i] = s;
}

extern "C" void kernel_launch(void* const* d_in, const int* in_sizes, int n_in,
                              void* d_out, int out_size, void* d_ws, size_t ws_size,
                              hipStream_t stream){
  const float* x      = (const float*)d_in[0];
  const int*   eidx   = (const int*)  d_in[1];
  const int*   etype  = (const int*)  d_in[2];
  const int*   batch  = (const int*)  d_in[3];
  const float* in_w   = (const float*)d_in[4];
  const float* in_b   = (const float*)d_in[5];
  const float* root_w = (const float*)d_in[6];
  const float* rel_w  = (const float*)d_in[7];
  const float* conv_b = (const float*)d_in[8];
  const float* ln_w   = (const float*)d_in[9];
  const float* ln_b   = (const float*)d_in[10];
  const float* cls_w1 = (const float*)d_in[11];
  const float* cls_b1 = (const float*)d_in[12];
  const float* cls_w2 = (const float*)d_in[13];
  const float* cls_b2 = (const float*)d_in[14];
  float* out = (float*)d_out;

  char* ws = (char*)d_ws;
  size_t off = 0;
  auto alloc = [&](size_t bytes) -> char* {
    char* p = ws + off; off += (bytes + 255) & ~(size_t)255; return p;
  };
  float*  hbuf    = (float*) alloc((size_t)NN * HID * 4);          // current h
  float*  gout    = (float*) alloc((size_t)NN * HID * 4);          // GEMM output
  float*  acc     = (float*) alloc((size_t)NN * NREL * HID * 4);   // per-rel mean msgs
  float*  cnt     = (float*) alloc((size_t)NREL * NN * 4);
  float*  pooled  = (float*) alloc((size_t)NGRAPH * HID * 4);
  float*  zbuf    = (float*) alloc((size_t)NGRAPH * HID * 4);
  __bf16* wt_in   = (__bf16*)alloc((size_t)HID * 768 * 2);         // [256][768]
  __bf16* wt_root = (__bf16*)alloc((size_t)3 * HID * HID * 2);     // 3 x [256][256]
  __bf16* wt_rel  = (__bf16*)alloc((size_t)3 * HID * 768 * 2);     // 3 x [256][768] (rels stacked on K)
  __bf16* wt_c1   = (__bf16*)alloc((size_t)HID * HID * 2);         // [256][256]

  const int* src = eidx;
  const int* dst = eidx + EE;

  // --- prep bf16 transposed weights ---
  k_transpose_bf16<<<(768 * HID + 255) / 256, 256, 0, stream>>>(in_w, wt_in, 768, 768, 0);
  for (int l = 0; l < 3; ++l)
    k_transpose_bf16<<<(HID * HID + 255) / 256, 256, 0, stream>>>(
        root_w + (size_t)l * HID * HID, wt_root + (size_t)l * HID * HID, HID, HID, 0);
  for (int l = 0; l < 3; ++l)
    for (int r = 0; r < 3; ++r)
      k_transpose_bf16<<<(HID * HID + 255) / 256, 256, 0, stream>>>(
          rel_w + (size_t)(l * 3 + r) * HID * HID, wt_rel + (size_t)l * HID * 768, HID, 768, r * HID);
  k_transpose_bf16<<<(HID * HID + 255) / 256, 256, 0, stream>>>(cls_w1, wt_c1, HID, HID, 0);

  // waves = mtiles * 4 N-groups; 8 waves per block
  const int gemmBlocks = ((NN / 16) * 4 + 7) / 8;     // 1563

  // --- input projection: h = x @ in_w + in_b ---
  k_wmma_gemm<<<gemmBlocks, 256, 0, stream>>>(x, 768, 768, wt_in, in_b, hbuf, NN, 0, 0);

  // --- RGCN layers ---
  for (int l = 0; l < 3; ++l){
    hipMemsetAsync(acc, 0, (size_t)NN * NREL * HID * 4, stream);
    hipMemsetAsync(cnt, 0, (size_t)NREL * NN * 4, stream);
    k_scatter<<<EE / 8, 256, 0, stream>>>(hbuf, src, dst, etype, acc, cnt);
    k_mean<<<(NN * NREL * HID / 4 + 255) / 256, 256, 0, stream>>>(acc, cnt);
    k_wmma_gemm<<<gemmBlocks, 256, 0, stream>>>(hbuf, HID, HID,
        wt_root + (size_t)l * HID * HID, conv_b + l * HID, gout, NN, 0, 0);
    k_wmma_gemm<<<gemmBlocks, 256, 0, stream>>>(acc, 768, 768,
        wt_rel + (size_t)l * HID * 768, nullptr, gout, NN, 1, 0);
    k_epilogue<<<(NN + 7) / 8, 256, 0, stream>>>(gout, hbuf, ln_w + l * HID, ln_b + l * HID);
  }

  // --- pool + classifier ---
  k_pool_init<<<(NGRAPH * HID + 255) / 256, 256, 0, stream>>>(pooled);
  k_pool<<<(NN + 7) / 8, 256, 0, stream>>>(hbuf, batch, pooled);
  // z = relu(pooled @ cls_w1 + b1) on the WMMA path (M=64 -> 16 waves)
  k_wmma_gemm<<<((NGRAPH / 16) * 4 + 7) / 8, 256, 0, stream>>>(
      pooled, HID, HID, wt_c1, cls_b1, zbuf, NGRAPH, 0, 1);
  k_cls2<<<1, 256, 0, stream>>>(zbuf, cls_w2, cls_b2, out);
}
